// TriLinear_Attn_17360257810685
// MI455X (gfx1250) — matrix operand order repaired and checked by reading, not compiled, not run on hardware
//
#include <hip/hip_runtime.h>

// MI455X / gfx1250: wave32, WMMA bf16 16x16x32 (f32 accum), TDM async tensor loads.
// Workspace (f32 unless noted):
//   qf 32K | kf 16K | rowmax 32K | q2c 8K | c2q 8M | Gf 16M | Gb 16M floats
//   + resultbf 32M bf16  => ~224 MiB total.

typedef __bf16 bf16_t;
typedef bf16_t v16bf __attribute__((ext_vector_type(16)));
typedef float  v8f   __attribute__((ext_vector_type(8)));
typedef float  f4    __attribute__((ext_vector_type(4)));
typedef unsigned int u32x4 __attribute__((ext_vector_type(4)));
typedef int          i32x4 __attribute__((ext_vector_type(4)));
typedef int          i32x8 __attribute__((ext_vector_type(8)));

#if __has_include(<hip/amd_detail/amd_gfx1250_TDM.h>)
#define TDM_6ARG 1
#else
#define TDM_6ARG 0
#endif

#define NEGV (-1e10f)

__device__ __forceinline__ v8f v8zero() {
  v8f r;
  #pragma unroll
  for (int i = 0; i < 8; ++i) r[i] = 0.f;
  return r;
}

// ---- TDM: async 2D tile load Global -> LDS (descriptor per CDNA5 ISA D#) ----
// elem_code: 0=1B,1=2B,2=4B. Optional LDS row padding: pad_amount DWORD-code every
// 2<<pad_interval_code DWORDs (see D# group1 bits [31:20]).
__device__ __forceinline__ void tdm_load_2d(unsigned int lds_byte_off, const void* gptr,
                                            unsigned int elem_code,
                                            unsigned int tensor_d0,
                                            unsigned int tile_d0, unsigned int tile_d1,
                                            unsigned int stride0,
                                            unsigned int pad_enable,
                                            unsigned int pad_interval_code,
                                            unsigned int pad_amount_code) {
  const unsigned long long ga = (unsigned long long)(uintptr_t)gptr;
  const unsigned int TD1 = 1u << 20;  // large tensor_dim1 (no OOB clipping needed)
  u32x4 g0;
  g0[0] = 1u;                                        // count=1 (valid user descriptor)
  g0[1] = lds_byte_off;                              // lds_addr
  g0[2] = (unsigned int)(ga & 0xffffffffu);          // global_addr[31:0]
  g0[3] = (unsigned int)((ga >> 32) & 0x01ffffffu) | 0x80000000u;  // addr[56:32] | type=2
  i32x8 g1;
  g1[0] = (int)((elem_code << 16) | (pad_enable << 20) |
                (pad_interval_code << 22) | (pad_amount_code << 25));
  g1[1] = (int)((tensor_d0 & 0xffffu) << 16);                       // tensor_dim0[15:0]
  g1[2] = (int)((tensor_d0 >> 16) | ((TD1 & 0xffffu) << 16));       // dim0 hi | dim1 lo
  g1[3] = (int)((TD1 >> 16) | ((tile_d0 & 0xffffu) << 16));         // dim1 hi | tile_dim0
  g1[4] = (int)(tile_d1 & 0xffffu);                                 // tile_dim1 (tile_dim2=0)
  g1[5] = (int)stride0;                                             // dim0_stride[31:0]
  g1[6] = 0;
  g1[7] = 0;
  i32x4 z4 = {0, 0, 0, 0};
#if TDM_6ARG
  i32x8 z8 = {0, 0, 0, 0, 0, 0, 0, 0};
  __builtin_amdgcn_tensor_load_to_lds(g0, g1, z4, z4, z8, 0);
#else
  __builtin_amdgcn_tensor_load_to_lds(g0, g1, z4, z4, 0);
#endif
}

__device__ __forceinline__ void tdm_wait() {
  __builtin_amdgcn_s_wait_tensorcnt((short)0);
}

// A-matrix fragment (bf16 16x32), source row-major [m][k] in LDS, ld mult of 8 elems.
__device__ __forceinline__ v16bf load_a16x32(const bf16_t* p, int ld) {
  const int l  = threadIdx.x & 31;
  const int m  = l & 15;
  const int kb = (l < 16) ? 0 : 8;
  const bf16_t* r0 = p + m * ld + kb;
  v16bf r;
  *(f4*)&r        = *(const f4*)(r0);
  *((f4*)&r + 1)  = *(const f4*)(r0 + 16);
  return r;
}

// B-matrix fragment (bf16 32x16), source stored [n][k] row-major in LDS.
__device__ __forceinline__ v16bf load_b32x16(const bf16_t* p, int ld) {
  const int l  = threadIdx.x & 31;
  const int n  = l & 15;
  const int kb = (l < 16) ? 0 : 8;
  const bf16_t* r0 = p + n * ld + kb;
  v16bf r;
  *(f4*)&r        = *(const f4*)(r0);
  *((f4*)&r + 1)  = *(const f4*)(r0 + 16);
  return r;
}

__device__ __forceinline__ v8f wmma_bf16(v16bf a, v16bf b, v8f c) {
  return __builtin_amdgcn_wmma_f32_16x16x32_bf16(false, a, false, b, (short)0, c,
                                                 false, false);
}

__device__ __forceinline__ float halfmax(float v) {
  #pragma unroll
  for (int m = 1; m < 16; m <<= 1) v = fmaxf(v, __shfl_xor(v, m, 32));
  return v;
}
__device__ __forceinline__ float halfsum(float v) {
  #pragma unroll
  for (int m = 1; m < 16; m <<= 1) v += __shfl_xor(v, m, 32);
  return v;
}

// ---------------- Kernel 0: qf = Q.query_w, kf = K.key_w ----------------
__global__ void qkf_kernel(const float* __restrict__ qg, const float* __restrict__ kg,
                           const float* __restrict__ qw, const float* __restrict__ kw,
                           float* __restrict__ qf, float* __restrict__ kf) {
  int r = blockIdx.x * blockDim.x + threadIdx.x;
  const int nq = 32 * 1024;
  if (r < nq) {
    const float* row = qg + (size_t)r * 256;
    float acc = 0.f;
    for (int d = 0; d < 256; ++d) acc += row[d] * qw[d];
    qf[r] = acc;
  } else if (r < nq + 32 * 512) {
    int rr = r - nq;
    const float* row = kg + (size_t)rr * 256;
    float acc = 0.f;
    for (int d = 0; d < 256; ++d) acc += row[d] * kw[d];
    kf[rr] = acc;
  }
}

// ---------------- Kernel 1: flash attention (c2q + rowmax) ----------------
__global__ void __launch_bounds__(128, 1)
attn_kernel(const float* __restrict__ qg, const float* __restrict__ kg,
            const unsigned char* __restrict__ qmask, const unsigned char* __restrict__ kmask,
            const float* __restrict__ dotw,
            const float* __restrict__ qf, const float* __restrict__ kf,
            float* __restrict__ c2q, float* __restrict__ rowmax_out) {
  extern __shared__ char smem[];
  const int LDQ = 264, LDK = 264, LDD = 72;
  bf16_t* Qs = (bf16_t*)smem;          // 64 x LDQ  (Q * dot_w, bf16)
  bf16_t* Kt = Qs + 64 * LDQ;          // 64 x LDK  (K tile, [k][d])
  bf16_t* Kd = Kt + 64 * LDK;          // 256 x LDD (K tile transposed, [d][k])
  bf16_t* Pl = Kd + 256 * LDD;         // 4 waves * 16 x 64 (P tile)

  const int b    = blockIdx.y;
  const int q0   = blockIdx.x * 64;
  const int tid  = threadIdx.x;
  const int wave = tid >> 5;
  const int lane = tid & 31;
  const int ln   = lane & 15;
  const int rh   = (lane >> 4) * 8;

  for (int idx = tid; idx < 64 * 256; idx += 128) {
    int r = idx >> 8, d = idx & 255;
    Qs[r * LDQ + d] = (bf16_t)(qg[((size_t)b * 1024 + q0 + r) * 256 + d] * dotw[d]);
  }

  float qfr[8]; int qm[8];
  #pragma unroll
  for (int j = 0; j < 8; ++j) {
    int row = wave * 16 + rh + j;
    qfr[j] = qf[b * 1024 + q0 + row];
    qm[j]  = qmask[b * 1024 + q0 + row];
  }

  float mrun[8], lrun[8];
  v8f o[16];
  #pragma unroll
  for (int j = 0; j < 8; ++j) { mrun[j] = -3.0e38f; lrun[j] = 0.f; }
  #pragma unroll
  for (int t = 0; t < 16; ++t) o[t] = v8zero();

  for (int k0 = 0; k0 < 512; k0 += 64) {
    __syncthreads();
    for (int idx = tid; idx < 64 * 256; idx += 128) {
      int k = idx >> 8, d = idx & 255;
      bf16_t v = (bf16_t)kg[((size_t)b * 512 + k0 + k) * 256 + d];
      Kt[k * LDK + d] = v;
      Kd[d * LDD + k] = v;
    }
    __syncthreads();

    v8f s[4];
    #pragma unroll
    for (int t = 0; t < 4; ++t) s[t] = v8zero();
    const bf16_t* Abase = Qs + wave * 16 * LDQ;
    for (int kc = 0; kc < 256; kc += 32) {
      v16bf a = load_a16x32(Abase + kc, LDQ);
      #pragma unroll
      for (int t = 0; t < 4; ++t) {
        v16bf bb = load_b32x16(Kt + (t * 16) * LDK + kc, LDK);
        s[t] = wmma_bf16(a, bb, s[t]);
      }
    }

    float kfv[4]; int kmv[4];
    #pragma unroll
    for (int t = 0; t < 4; ++t) {
      kfv[t] = kf[b * 512 + k0 + t * 16 + ln];
      kmv[t] = kmask[b * 512 + k0 + t * 16 + ln];
    }
    #pragma unroll
    for (int t = 0; t < 4; ++t)
      #pragma unroll
      for (int j = 0; j < 8; ++j) {
        float v = s[t][j] + qfr[j] + kfv[t];
        if (qm[j] | kmv[t]) v = NEGV;
        s[t][j] = v;
      }

    float mnew[8], scale[8];
    #pragma unroll
    for (int j = 0; j < 8; ++j) {
      float ml = fmaxf(fmaxf(s[0][j], s[1][j]), fmaxf(s[2][j], s[3][j]));
      ml = halfmax(ml);
      mnew[j]  = fmaxf(mrun[j], ml);
      scale[j] = __expf(mrun[j] - mnew[j]);
      mrun[j]  = mnew[j];
    }
    bf16_t* Pw = Pl + wave * 16 * 64;
    float lsum[8];
    #pragma unroll
    for (int j = 0; j < 8; ++j) lsum[j] = 0.f;
    #pragma unroll
    for (int t = 0; t < 4; ++t)
      #pragma unroll
      for (int j = 0; j < 8; ++j) {
        float p = __expf(s[t][j] - mnew[j]);
        lsum[j] += p;
        Pw[(rh + j) * 64 + t * 16 + ln] = (bf16_t)p;
      }
    #pragma unroll
    for (int j = 0; j < 8; ++j) lrun[j] = lrun[j] * scale[j] + halfsum(lsum[j]);
    #pragma unroll
    for (int t = 0; t < 16; ++t)
      #pragma unroll
      for (int j = 0; j < 8; ++j) o[t][j] *= scale[j];

    __syncthreads();

    for (int kc = 0; kc < 64; kc += 32) {
      v16bf a = load_a16x32(Pw + kc, 64);
      #pragma unroll
      for (int dt = 0; dt < 16; ++dt) {
        v16bf bb = load_b32x16(Kd + (dt * 16) * LDD + kc, LDD);
        o[dt] = wmma_bf16(a, bb, o[dt]);
      }
    }
  }

  float rinv[8];
  #pragma unroll
  for (int j = 0; j < 8; ++j) rinv[j] = 1.0f / lrun[j];
  #pragma unroll
  for (int dt = 0; dt < 16; ++dt)
    #pragma unroll
    for (int j = 0; j < 8; ++j) {
      int row = wave * 16 + rh + j;
      c2q[((size_t)b * 1024 + q0 + row) * 256 + dt * 16 + ln] = o[dt][j] * rinv[j];
    }
  if (ln == 0)
    #pragma unroll
    for (int j = 0; j < 8; ++j) {
      int row = wave * 16 + rh + j;
      rowmax_out[b * 1024 + q0 + row] = mrun[j];
    }
}

// ---------------- Kernel 2: q2c = softmax_Tq(rowmax) . Q ----------------
__global__ void q2c_kernel(const float* __restrict__ qg, const float* __restrict__ rowmax,
                           float* __restrict__ q2c) {
  __shared__ float w[1024];
  __shared__ float red[256];
  const int b = blockIdx.x, tid = threadIdx.x;
  float mx = -3e38f;
  for (int q = tid; q < 1024; q += 256) mx = fmaxf(mx, rowmax[b * 1024 + q]);
  red[tid] = mx; __syncthreads();
  for (int s = 128; s > 0; s >>= 1) { if (tid < s) red[tid] = fmaxf(red[tid], red[tid + s]); __syncthreads(); }
  mx = red[0]; __syncthreads();
  float sm = 0.f;
  for (int q = tid; q < 1024; q += 256) { float e = __expf(rowmax[b * 1024 + q] - mx); w[q] = e; sm += e; }
  red[tid] = sm; __syncthreads();
  for (int s = 128; s > 0; s >>= 1) { if (tid < s) red[tid] += red[tid + s]; __syncthreads(); }
  sm = red[0]; __syncthreads();
  const float inv = 1.f / sm;
  const int d = tid;
  float acc = 0.f;
  for (int q = 0; q < 1024; ++q) acc += w[q] * qg[((size_t)b * 1024 + q) * 256 + d];
  q2c[b * 256 + d] = acc * inv;
}

// ---------------- Kernel 3: assemble result features (bf16) ----------------
__global__ void assemble_kernel(const float* __restrict__ qg, const float* __restrict__ c2q,
                                const float* __restrict__ q2c,
                                const unsigned char* __restrict__ qmask,
                                bf16_t* __restrict__ resbf) {
  const int bq = blockIdx.x;   // 32*1024
  const int d  = threadIdx.x;  // 256
  const int b  = bq >> 10;
  const size_t base = (size_t)bq * 256 + d;
  const float qv = qg[base];
  const float cv = c2q[base];
  const float gv = q2c[b * 256 + d];
  const float m  = qmask[bq] ? 0.f : 1.f;
  bf16_t* out = resbf + (size_t)bq * 1024;
  out[d]       = (bf16_t)(qv * m);
  out[256 + d] = (bf16_t)(cv * m);
  out[512 + d] = (bf16_t)(qv * cv * m);
  out[768 + d] = (bf16_t)(qv * gv * m);
}

// ---------------- Kernel 4: G = result @ W_ih^T + b_ih + b_hh ----------------
// A tiles staged by TDM (tensor_load_to_lds) with LDS padding to ld=72, double buffered.
__global__ void __launch_bounds__(128, 1)
ggemm_kernel(const bf16_t* __restrict__ resbf,
             const float* __restrict__ w_ih_f, const float* __restrict__ b_ih_f,
             const float* __restrict__ b_hh_f,
             const float* __restrict__ w_ih_b, const float* __restrict__ b_ih_b,
             const float* __restrict__ b_hh_b,
             float* __restrict__ Gf, float* __restrict__ Gb) {
  extern __shared__ char smem[];
  // LDS layout (dynamic only => base offset 0):
  //   As0 @ 0      (64x72 bf16 = 9216 B)   As1 @ 9216
  //   Bs0 @ 18432  (64x72 bf16)            Bs1 @ 27648     total 36864 B
  bf16_t* sm = (bf16_t*)smem;
  const int r0   = blockIdx.x * 64;
  const int cblk = blockIdx.y;            // 0..15 (8 fwd, 8 bwd)
  const int dir  = cblk >> 3;
  const int j0   = (cblk & 7) * 64;
  const float* w  = dir ? w_ih_b : w_ih_f;
  const float* bi = dir ? b_ih_b : b_ih_f;
  const float* bh = dir ? b_hh_b : b_hh_f;
  float* G = dir ? Gb : Gf;
  const int tid = threadIdx.x, wave = tid >> 5, lane = tid & 31, ln = lane & 15, rh = (lane >> 4) * 8;

  v8f acc[4];
  #pragma unroll
  for (int t = 0; t < 4; ++t) acc[t] = v8zero();

  // prologue: TDM chunk 0 -> As0; fill Bs0
  if (wave == 0)
    tdm_load_2d(0u, resbf + (size_t)r0 * 1024, 1u, 1024u, 64u, 64u, 1024u, 1u, 4u, 3u);
  for (int idx = tid; idx < 64 * 64; idx += 128) {
    int j = idx >> 6, k = idx & 63;
    sm[18432 / 2 + j * 72 + k] = (bf16_t)w[(size_t)(j0 + j) * 1024 + k];
  }

  for (int k0 = 0; k0 < 1024; k0 += 64) {
    const int buf = (k0 >> 6) & 1;
    if (wave == 0) tdm_wait();      // TDM for this chunk complete (wave0 issued it)
    __syncthreads();                // publish As[buf]; all waves done with other buffer
    if (k0 + 64 < 1024) {
      if (wave == 0)
        tdm_load_2d((unsigned)(buf ^ 1) * 9216u,
                    resbf + (size_t)r0 * 1024 + (k0 + 64), 1u, 1024u, 64u, 64u, 1024u,
                    1u, 4u, 3u);
      for (int idx = tid; idx < 64 * 64; idx += 128) {
        int j = idx >> 6, k = idx & 63;
        sm[18432 / 2 + (buf ^ 1) * 4608 + j * 72 + k] =
            (bf16_t)w[(size_t)(j0 + j) * 1024 + (k0 + 64) + k];
      }
    }
    const bf16_t* As = sm + buf * 4608;
    const bf16_t* Bs = sm + 18432 / 2 + buf * 4608;
    const bf16_t* Ab = As + wave * 16 * 72;
    for (int kc = 0; kc < 64; kc += 32) {
      v16bf a = load_a16x32(Ab + kc, 72);
      #pragma unroll
      for (int t = 0; t < 4; ++t) {
        v16bf bb = load_b32x16(Bs + (t * 16) * 72 + kc, 72);
        acc[t] = wmma_bf16(a, bb, acc[t]);
      }
    }
  }
  #pragma unroll
  for (int t = 0; t < 4; ++t)
    #pragma unroll
    for (int j = 0; j < 8; ++j) {
      int row = r0 + wave * 16 + rh + j;
      int col = j0 + t * 16 + ln;
      G[(size_t)row * 512 + col] = acc[t][j] + bi[col] + bh[col];
    }
}

// ---------------- Kernel 5: LSTM recurrence (32 steps, per-direction) ----------------
// W_hh resident in LDS; per-step gate preactivation tile staged by TDM.
__global__ void __launch_bounds__(128, 1)
lstm_kernel(const float* __restrict__ Gf, const float* __restrict__ Gb,
            const float* __restrict__ whh_f, const float* __restrict__ whh_b,
            const unsigned char* __restrict__ qmask,
            float* __restrict__ out) {
  extern __shared__ char smem[];
  // LDS layout (dynamic only): Whh @0 (512x128 bf16 = 131072 B)
  //   Hs @131072 (64x128 bf16 = 16384 B);  Gs @147456 (64x512 f32 = 131072 B)
  bf16_t* Whh = (bf16_t*)smem;
  bf16_t* Hs  = Whh + 512 * 128;
  float*  Gs  = (float*)(smem + 147456);
  const int dir = blockIdx.y;
  const int n0  = blockIdx.x * 64;
  const float* G  = dir ? Gb : Gf;
  const float* wg = dir ? whh_b : whh_f;
  const int tid = threadIdx.x, wave = tid >> 5, lane = tid & 31, ln = lane & 15, rh = (lane >> 4) * 8;

  for (int idx = tid; idx < 512 * 128; idx += 128) Whh[idx] = (bf16_t)wg[idx];

  v8f h[8], c[8];
  #pragma unroll
  for (int d = 0; d < 8; ++d) { h[d] = v8zero(); c[d] = v8zero(); }

  for (int step = 0; step < 32; ++step) {
    const int t = dir ? (31 - step) : step;
    __syncthreads();                       // prev step's reads of Gs/Hs complete
    if (wave == 0)                          // stage G[t] rows n0..n0+63 (64 x 512 f32)
      tdm_load_2d(147456u, G + ((size_t)t * 1024 + n0) * 512, 2u, 512u, 512u, 64u,
                  512u, 0u, 0u, 0u);
    #pragma unroll
    for (int d = 0; d < 8; ++d)
      #pragma unroll
      for (int j = 0; j < 8; ++j)
        Hs[(wave * 16 + rh + j) * 128 + d * 16 + ln] = (bf16_t)h[d][j];
    if (wave == 0) tdm_wait();
    __syncthreads();                       // publish Hs + Gs

    const bf16_t* Ab = Hs + wave * 16 * 128;
    for (int d = 0; d < 8; ++d) {          // hidden col tile
      v8f g4[4];
      #pragma unroll
      for (int gi = 0; gi < 4; ++gi) {     // i,f,g,o
        v8f gv;
        #pragma unroll
        for (int j = 0; j < 8; ++j)
          gv[j] = Gs[(wave * 16 + rh + j) * 512 + gi * 128 + d * 16 + ln];
        g4[gi] = gv;
      }
      for (int kc = 0; kc < 128; kc += 32) {
        v16bf a = load_a16x32(Ab + kc, 128);
        #pragma unroll
        for (int gi = 0; gi < 4; ++gi) {
          v16bf bb = load_b32x16(Whh + (gi * 128 + d * 16) * 128 + kc, 128);
          g4[gi] = wmma_bf16(a, bb, g4[gi]);
        }
      }
      #pragma unroll
      for (int j = 0; j < 8; ++j) {
        float ig = 1.f / (1.f + __expf(-g4[0][j]));
        float fg = 1.f / (1.f + __expf(-g4[1][j]));
        float gg = tanhf(g4[2][j]);
        float og = 1.f / (1.f + __expf(-g4[3][j]));
        float cn = fg * c[d][j] + ig * gg;
        c[d][j] = cn;
        h[d][j] = og * tanhf(cn);
      }
    }
    #pragma unroll
    for (int d = 0; d < 8; ++d)
      #pragma unroll
      for (int j = 0; j < 8; ++j) {
        int row = n0 + wave * 16 + rh + j;
        float v = qmask[t * 1024 + row] ? 0.f : h[d][j];
        out[((size_t)t * 1024 + row) * 256 + dir * 128 + d * 16 + ln] = v;
      }
  }
}

extern "C" void kernel_launch(void* const* d_in, const int* in_sizes, int n_in,
                              void* d_out, int out_size, void* d_ws, size_t ws_size,
                              hipStream_t stream) {
  const float* qg   = (const float*)d_in[0];
  const float* kg   = (const float*)d_in[1];
  const unsigned char* qmask = (const unsigned char*)d_in[2];
  const unsigned char* kmask = (const unsigned char*)d_in[3];
  const float* qw   = (const float*)d_in[4];
  const float* kw   = (const float*)d_in[5];
  const float* dotw = (const float*)d_in[6];
  const float* w_ih_f = (const float*)d_in[7];
  const float* w_hh_f = (const float*)d_in[8];
  const float* b_ih_f = (const float*)d_in[9];
  const float* b_hh_f = (const float*)d_in[10];
  const float* w_ih_b = (const float*)d_in[11];
  const float* w_hh_b = (const float*)d_in[12];
  const float* b_ih_b = (const float*)d_in[13];
  const float* b_hh_b = (const float*)d_in[14];
  float* out = (float*)d_out;

  float* ws = (float*)d_ws;
  size_t off = 0;
  float* qf     = ws + off; off += 32 * 1024;
  float* kf     = ws + off; off += 32 * 512;
  float* rowmax = ws + off; off += 32 * 1024;
  float* q2c    = ws + off; off += 32 * 256;
  float* c2q    = ws + off; off += (size_t)32 * 1024 * 256;
  float* Gf     = ws + off; off += (size_t)32 * 1024 * 512;
  float* Gb     = ws + off; off += (size_t)32 * 1024 * 512;
  bf16_t* resbf = (bf16_t*)(ws + off);   // 32*1024*1024 bf16

  // 0) row dot products
  qkf_kernel<<<192, 256, 0, stream>>>(qg, kg, qw, kw, qf, kf);

  // 1) flash attention
  const int attn_smem = (64 * 264 + 64 * 264 + 256 * 72 + 4 * 16 * 64) * 2;
  attn_kernel<<<dim3(16, 32), 128, attn_smem, stream>>>(qg, kg, qmask, kmask, dotw,
                                                        qf, kf, c2q, rowmax);

  // 2) q2c
  q2c_kernel<<<32, 256, 0, stream>>>(qg, rowmax, q2c);

  // 3) feature assembly (bf16)
  assemble_kernel<<<32 * 1024, 256, 0, stream>>>(qg, c2q, q2c, qmask, resbf);

  // 4) LSTM input GEMMs (both directions), TDM-staged A tiles, double buffered
  const int ggemm_smem = 36864;
  ggemm_kernel<<<dim3(512, 16), 128, ggemm_smem, stream>>>(resbf, w_ih_f, b_ih_f, b_hh_f,
                                                           w_ih_b, b_ih_b, b_hh_b, Gf, Gb);

  // 5) recurrence (both directions), TDM-staged gate tiles, writes masked output
  const int lstm_smem = 147456 + 131072;   // Whh + Hs + Gs
  lstm_kernel<<<dim3(16, 2), 128, lstm_smem, stream>>>(Gf, Gb, w_hh_f, w_hh_b, qmask, out);
}
